// PLEv3_74062416052344
// MI455X (gfx1250) — matrix operand order repaired and checked by reading, not compiled, not run on hardware
//
#include <hip/hip_runtime.h>
#include <hip/hip_bf16.h>
#include <stdint.h>

typedef __attribute__((ext_vector_type(16))) _Float16 v16h;
typedef __attribute__((ext_vector_type(8)))  float    v8f;

#define DEV __device__ __forceinline__

static constexpr int Bn = 32768;

// ---------------- device helpers ----------------

DEV float gelu_f(float x) { return 0.5f * x * (1.0f + erff(x * 0.70710678118f)); }

DEV float wred(float v) {
#pragma unroll
  for (int m = 16; m > 0; m >>= 1) v += __shfl_xor(v, m, 32);
  return v;
}

// A fragment (16x32 f16) from LDS, row-major with given stride (halfs).
// ISA layout: lanes 0-15 rows M=0..15 K-base 0, lanes 16-31 same rows K-base 8;
// elements 0..7 -> K=base..base+7, elements 8..15 -> K=base+16..base+23.
// mt is uniform per wave -> folded into the address, no per-element selects.
DEV v16h load_a_frag(const _Float16* base, int stride, int k0, int lane, int mt) {
  int m  = mt * 16 + (lane & 15);
  int kb = k0 + ((lane >> 4) << 3);
  const _Float16* p = base + m * stride + kb;
  v16h a;
#pragma unroll
  for (int j = 0; j < 8; ++j) a[j] = p[j];
#pragma unroll
  for (int j = 0; j < 8; ++j) a[8 + j] = p[16 + j];
  return a;
}

// B fragment (32x16 f16) straight from global weights (K x N row-major).
// ISA layout: lane l holds column N=l%16; elements 0..7 -> K=8*(l/16)+0..7, elements 8..15 -> K=16+8*(l/16)+0..7.
DEV v16h load_b_frag_g(const _Float16* __restrict__ W, int ndim, int k0, int n0, int lane) {
  const _Float16* p = W + (size_t)(k0 + ((lane >> 4) << 3)) * ndim + n0 + (lane & 15);
  v16h b;
#pragma unroll
  for (int j = 0; j < 8; ++j) b[j] = p[(size_t)j * ndim];
#pragma unroll
  for (int j = 0; j < 8; ++j) b[8 + j] = p[(size_t)(16 + j) * ndim];
  return b;
}

DEV v8f wmma16(v16h a, v16h b, v8f c) {
  return __builtin_amdgcn_wmma_f32_16x16x32_f16(false, a, false, b, (short)0, c, false, false);
}

// ---------------- generic WMMA GEMM:  Out[b, :] = act(A[b, 0:K] @ W + bias) ----------------
// 256 threads (8 waves), 32 rows per block, NDIM columns. A f16 (lda halfs).
// Waves 0-3 own m-tile 0, waves 4-7 own m-tile 1: one A-fragment per wave per K-step.
// Double-buffered A tile (one barrier per K-step) + weight prefetch for next K-step.
template <int KDIM, int NDIM, int ACT>
__global__ __launch_bounds__(256) void ple_gemm_k(const _Float16* __restrict__ A, int lda,
                                                  const _Float16* __restrict__ W,
                                                  const float* __restrict__ bias,
                                                  _Float16* __restrict__ Out, int ldo) {
  constexpr int NT  = NDIM / 16;
  constexpr int TPW = (NT + 3) / 4;
  __shared__ __align__(16) _Float16 As[2][32][40];
  const int tid = threadIdx.x, wave = tid >> 5, lane = tid & 31;
  const int mt = wave >> 2, wn = wave & 3;
  const int rb = blockIdx.x;
  v8f acc[TPW] = {};
  const int r = tid >> 3, c4 = (tid & 7) * 4;
  const _Float16* arow = A + (size_t)(rb * 32 + r) * lda + c4;
  *(uint64_t*)(&As[0][r][c4]) = *(const uint64_t*)arow;  // prologue: K-step 0
  __syncthreads();
  int cur = 0;
  for (int k0 = 0; k0 < KDIM; k0 += 32) {
    if (k0 + 32 < KDIM)  // stage next K-step into the other buffer (overlaps with WMMAs)
      *(uint64_t*)(&As[cur ^ 1][r][c4]) = *(const uint64_t*)(arow + k0 + 32);
    v16h a = load_a_frag(&As[cur][0][0], 40, 0, lane, mt);
#pragma unroll
    for (int j = 0; j < TPW; ++j) {
      int nt = wn + j * 4;
      if (nt < NT) {
        if (k0 + 32 < KDIM)
          __builtin_prefetch(W + (size_t)(k0 + 32 + lane) * NDIM + nt * 16, 0, 1);
        acc[j] = wmma16(a, load_b_frag_g(W, NDIM, k0, nt * 16, lane), acc[j]);
      }
    }
    __syncthreads();
    cur ^= 1;
  }
#pragma unroll
  for (int j = 0; j < TPW; ++j) {
    int nt = wn + j * 4;
    if (nt < NT) {
      int nc    = nt * 16 + (lane & 15);
      int rbase = rb * 32 + mt * 16 + ((lane >> 4) << 3);
#pragma unroll
      for (int i = 0; i < 8; ++i) {
        float v = acc[j][i] + bias[nc];
        if (ACT == 1) v = gelu_f(v);
        Out[(size_t)(rbase + i) * ldo + nc] = (_Float16)v;
      }
    }
  }
}

// ---------------- expert-stack GEMM phases operating LDS -> LDS ----------------
// 32 rows (2 m-tiles), NDIM cols; exact 8-wave tiling (NDIM/16 divisible by 4).

template <int KDIM, int NDIM>
DEV void ple_phase_f16(const _Float16* Al, int lda, const _Float16* __restrict__ Wg,
                       const float* __restrict__ bias, _Float16* Ho, int ldo,
                       int wave, int lane) {
  constexpr int TPW = NDIM / 64;
  const int mt = wave >> 2, wn = wave & 3;
  v8f acc[TPW] = {};
  for (int k0 = 0; k0 < KDIM; k0 += 32) {
    v16h a = load_a_frag(Al, lda, k0, lane, mt);
#pragma unroll
    for (int j = 0; j < TPW; ++j) {
      int nt = wn + j * 4;
      if (k0 + 32 < KDIM)
        __builtin_prefetch(Wg + (size_t)(k0 + 32 + lane) * NDIM + nt * 16, 0, 1);
      acc[j] = wmma16(a, load_b_frag_g(Wg, NDIM, k0, nt * 16, lane), acc[j]);
    }
  }
#pragma unroll
  for (int j = 0; j < TPW; ++j) {
    int nt = wn + j * 4;
    int nc = nt * 16 + (lane & 15);
    int rl = mt * 16 + ((lane >> 4) << 3);
#pragma unroll
    for (int i = 0; i < 8; ++i)
      Ho[(rl + i) * ldo + nc] = (_Float16)gelu_f(acc[j][i] + bias[nc]);
  }
}

template <int KDIM, int NDIM>
DEV void ple_phase_f32(const _Float16* Al, int lda, const _Float16* __restrict__ Wg,
                       const float* __restrict__ bias, float* Ho, int ldo,
                       int wave, int lane) {
  constexpr int TPW = NDIM / 64;
  const int mt = wave >> 2, wn = wave & 3;
  v8f acc[TPW] = {};
  for (int k0 = 0; k0 < KDIM; k0 += 32) {
    v16h a = load_a_frag(Al, lda, k0, lane, mt);
#pragma unroll
    for (int j = 0; j < TPW; ++j) {
      int nt = wn + j * 4;
      if (k0 + 32 < KDIM)
        __builtin_prefetch(Wg + (size_t)(k0 + 32 + lane) * NDIM + nt * 16, 0, 1);
      acc[j] = wmma16(a, load_b_frag_g(Wg, NDIM, k0, nt * 16, lane), acc[j]);
    }
  }
#pragma unroll
  for (int j = 0; j < TPW; ++j) {
    int nt = wn + j * 4;
    int nc = nt * 16 + (lane & 15);
    int rl = mt * 16 + ((lane >> 4) << 3);
#pragma unroll
    for (int i = 0; i < 8; ++i) Ho[(rl + i) * ldo + nc] = acc[j][i] + bias[nc];
  }
}

// ---------------- fused 3-layer expert stack + LayerNorm ----------------
// grid (B/32, E). 32 rows/block doubles B-operand reuse; h1/h2 never touch HBM.
// f32 eo tile aliases the (dead-after-phase-2) h1 buffer: ~71 KB static LDS.
__global__ __launch_bounds__(256) void ple_expert_k(
    const float* __restrict__ feat, const int* __restrict__ idxm,
    const _Float16* __restrict__ W1, const float* __restrict__ b1,
    const _Float16* __restrict__ W2, const float* __restrict__ b2,
    const _Float16* __restrict__ W3, const float* __restrict__ b3,
    const float* __restrict__ lg, const float* __restrict__ lb,
    _Float16* __restrict__ eoacct) {
  __shared__ __align__(16) char smem[4608 + 33280 + 33280];
  _Float16* xsS = (_Float16*)smem;                   // 32 x 72
  _Float16* h1S = (_Float16*)(smem + 4608);          // 32 x 520
  _Float16* h2S = (_Float16*)(smem + 4608 + 33280);  // 32 x 520
  float*    eoS = (float*)(smem + 4608);             // 32 x 256 (aliases h1S)
  const int tid = threadIdx.x, wave = tid >> 5, lane = tid & 31;
  const int rb = blockIdx.x, e = blockIdx.y;

  {  // gather + f32->f16 convert the 32x64 xs tile (8 elems/thread)
    int r = tid >> 3, c8 = (tid & 7) * 8;
#pragma unroll
    for (int j = 0; j < 8; ++j) {
      int c = c8 + j;
      int f = idxm[e * 64 + c];
      xsS[r * 72 + c] = (_Float16)feat[(size_t)(rb * 32 + r) * 256 + f];
    }
  }
  __syncthreads();
  // h1 = gelu(xs @ W1 + b1)            K=64  N=512
  ple_phase_f16<64, 512>(xsS, 72, W1 + (size_t)e * 32768, b1 + e * 512, h1S, 520, wave, lane);
  __syncthreads();
  // h2 = gelu(h1 @ W2 + b2)            K=512 N=512
  ple_phase_f16<512, 512>(h1S, 520, W2 + (size_t)e * 262144, b2 + e * 512, h2S, 520, wave, lane);
  __syncthreads();
  // eo_raw = h2 @ W3 + b3 (f32)        K=512 N=256  (writes alias h1S; h1 is dead)
  ple_phase_f32<512, 256>(h2S, 520, W3 + (size_t)e * 131072, b3 + e * 256, eoS, 256, wave, lane);
  __syncthreads();

  // LayerNorm over 256 cols; 4 rows per wave
#pragma unroll
  for (int rr = 0; rr < 4; ++rr) {
    int r = wave * 4 + rr;
    float s = 0.f, sq = 0.f;
#pragma unroll
    for (int j = 0; j < 8; ++j) {
      float v = eoS[r * 256 + j * 32 + lane];
      s += v;
      sq += v * v;
    }
    s = wred(s);
    sq = wred(sq);
    float mean = s * (1.f / 256.f);
    float var  = sq * (1.f / 256.f) - mean * mean;
    float rs   = rsqrtf(var + 1e-5f);
    size_t row = (size_t)(rb * 32 + r);
#pragma unroll
    for (int j = 0; j < 8; ++j) {
      int n   = j * 32 + lane;
      float v = (eoS[r * 256 + n] - mean) * rs * lg[e * 256 + n] + lb[e * 256 + n];
      eoacct[row * 1152 + e * 256 + n] = (_Float16)v;
    }
  }
}

// ---------------- fusion layer: fused = LN(gelu([gated|acct] @ fW + fB)) ----------------
// grid B/16. Block covers all 512 output columns -> in-block LayerNorm. Double-buffered A.
__global__ __launch_bounds__(256) void ple_fuse_ln_k(
    const _Float16* __restrict__ A /*gacct Bx384*/, const _Float16* __restrict__ W /*384x512*/,
    const float* __restrict__ bias, const float* __restrict__ lg, const float* __restrict__ lb,
    _Float16* __restrict__ seq, _Float16* __restrict__ pin) {
  __shared__ __align__(16) _Float16 As[2][16][40];
  __shared__ float accS[16][512];
  const int tid = threadIdx.x, wave = tid >> 5, lane = tid & 31;
  const int rb = blockIdx.x;
  v8f acc[4] = {};
  const int r = tid >> 4, c2 = (tid & 15) * 2;
  const _Float16* arow = A + (size_t)(rb * 16 + r) * 384 + c2;
  *(uint32_t*)(&As[0][r][c2]) = *(const uint32_t*)arow;
  __syncthreads();
  int cur = 0;
  for (int k0 = 0; k0 < 384; k0 += 32) {
    if (k0 + 32 < 384)
      *(uint32_t*)(&As[cur ^ 1][r][c2]) = *(const uint32_t*)(arow + k0 + 32);
    v16h a = load_a_frag(&As[cur][0][0], 40, 0, lane, 0);
#pragma unroll
    for (int j = 0; j < 4; ++j) {
      int nt = wave + j * 8;
      if (k0 + 32 < 384)
        __builtin_prefetch(W + (size_t)(k0 + 32 + lane) * 512 + nt * 16, 0, 1);
      acc[j] = wmma16(a, load_b_frag_g(W, 512, k0, nt * 16, lane), acc[j]);
    }
    __syncthreads();
    cur ^= 1;
  }
#pragma unroll
  for (int j = 0; j < 4; ++j) {
    int nt = wave + j * 8;
    int nc = nt * 16 + (lane & 15), rl = ((lane >> 4) << 3);
#pragma unroll
    for (int i = 0; i < 8; ++i) accS[rl + i][nc] = gelu_f(acc[j][i] + bias[nc]);
  }
  __syncthreads();
#pragma unroll
  for (int rr = 0; rr < 2; ++rr) {
    int rw = wave * 2 + rr;
    float s = 0.f, sq = 0.f;
#pragma unroll
    for (int j = 0; j < 16; ++j) {
      float v = accS[rw][j * 32 + lane];
      s += v;
      sq += v * v;
    }
    s = wred(s);
    sq = wred(sq);
    float mean = s * (1.f / 512.f);
    float var  = sq * (1.f / 512.f) - mean * mean;
    float rs   = rsqrtf(var + 1e-5f);
    size_t row = (size_t)(rb * 16 + rw);
#pragma unroll
    for (int j = 0; j < 16; ++j) {
      int n   = j * 32 + lane;
      float v = (accS[rw][n] - mean) * rs * lg[n] + lb[n];
      seq[row * 896 + n] = (_Float16)v;
      pin[row * 544 + n] = (_Float16)v;
    }
    pin[row * 544 + 512 + lane] = (_Float16)0.f;  // one-hot pad (overwritten by tails)
  }
}

// ---------------- small VALU kernels ----------------

__global__ void ple_pack_f16_k(const float* __restrict__ src, _Float16* __restrict__ dst, int n) {
  int i = blockIdx.x * 256 + threadIdx.x;
  if (i < n) dst[i] = (_Float16)src[i];
}

__global__ void ple_packmm_k(const float* __restrict__ maeW1, const float* __restrict__ mfeW1,
                             const float* __restrict__ maeB1, const float* __restrict__ mfeB1,
                             _Float16* __restrict__ dst, float* __restrict__ mmB) {
  int i = blockIdx.x * 256 + threadIdx.x;
  if (i < 544 * 128) {
    int k = i >> 7, j = i & 127;
    float v = 0.f;
    if (k < 526) v = (j < 64) ? maeW1[k * 64 + j] : mfeW1[k * 64 + (j - 64)];
    dst[i] = (_Float16)v;
  }
  if (i < 128) mmB[i] = (i < 64) ? maeB1[i] : mfeB1[i - 64];
}

__global__ void ple_acct_k(const float* __restrict__ as, const float* __restrict__ aW,
                           const float* __restrict__ aB, _Float16* __restrict__ eoacct,
                           _Float16* __restrict__ gacct) {
  int t = blockIdx.x * 256 + threadIdx.x;  // Bn*32 threads
  int b = t >> 5, j0 = (t & 31) * 4;
  float av[8];
#pragma unroll
  for (int k = 0; k < 8; ++k) av[k] = as[(size_t)b * 8 + k];
#pragma unroll
  for (int j = 0; j < 4; ++j) {
    int n   = j0 + j;
    float v = aB[n];
#pragma unroll
    for (int k = 0; k < 8; ++k) v += av[k] * aW[k * 128 + n];
    v = gelu_f(v);
    eoacct[(size_t)b * 1152 + 1024 + n] = (_Float16)v;
    gacct[(size_t)b * 384 + 256 + n]    = (_Float16)v;
  }
}

// scores -> softmax -> gated mix. One wave per row.
__global__ __launch_bounds__(256) void ple_gate_k(const _Float16* __restrict__ qh,
                                                  const _Float16* __restrict__ kkh,
                                                  const _Float16* __restrict__ eoacct,
                                                  _Float16* __restrict__ gacct,
                                                  float* __restrict__ out_gw) {
  int wave = threadIdx.x >> 5, lane = threadIdx.x & 31;
  size_t b = (size_t)blockIdx.x * 8 + wave;
  float qv[8];
#pragma unroll
  for (int j = 0; j < 8; ++j) qv[j] = (float)qh[b * 256 + lane * 8 + j];
  float sc[4];
#pragma unroll
  for (int e = 0; e < 4; ++e) {
    float p = 0.f;
#pragma unroll
    for (int j = 0; j < 8; ++j) p += qv[j] * (float)kkh[b * 1024 + e * 256 + lane * 8 + j];
    sc[e] = wred(p) * 0.0625f;  // 1/sqrt(256)
  }
  float mx = fmaxf(fmaxf(sc[0], sc[1]), fmaxf(sc[2], sc[3]));
  float w[4], se = 0.f;
#pragma unroll
  for (int e = 0; e < 4; ++e) { w[e] = expf(sc[e] - mx); se += w[e]; }
  float inv = 1.f / se;
#pragma unroll
  for (int e = 0; e < 4; ++e) w[e] *= inv;
  if (lane < 4) out_gw[b * 4 + lane] = w[lane];
#pragma unroll
  for (int j = 0; j < 8; ++j) {
    int n   = j * 32 + lane;
    float g = 0.f;
#pragma unroll
    for (int e = 0; e < 4; ++e) g += w[e] * (float)eoacct[b * 1152 + e * 256 + n];
    gacct[b * 384 + n] = (_Float16)g;
  }
}

// decision head tail: logits = hid @ W2 + b2, argmax, embedding feedback, one-hot. 1 wave/row.
__global__ __launch_bounds__(256) void ple_head_tail_k(
    const _Float16* __restrict__ hid, const float* __restrict__ W2, const float* __restrict__ b2,
    int NL, const float* __restrict__ Emb, _Float16* __restrict__ seq, int seq_off,
    _Float16* __restrict__ pin, int oh_off, int* __restrict__ idxbuf, int slot,
    float* __restrict__ out_logits, float* __restrict__ out_sidx) {
  int wave = threadIdx.x >> 5, lane = threadIdx.x & 31;
  size_t b = (size_t)blockIdx.x * 8 + wave;
  float part[4] = {0.f, 0.f, 0.f, 0.f};
#pragma unroll
  for (int j = 0; j < 8; ++j) {
    float h = (float)hid[b * 256 + lane * 8 + j];
    for (int l = 0; l < NL; ++l) part[l] += h * W2[(lane * 8 + j) * NL + l];
  }
  for (int l = 0; l < NL; ++l) part[l] = wred(part[l]) + b2[l];
  int best = 0;
  for (int l = 1; l < NL; ++l)
    if (part[l] > part[best]) best = l;
  if (lane < NL) out_logits[b * NL + lane] = part[lane];
  if (Emb) {
#pragma unroll
    for (int j = 0; j < 4; ++j) {
      int n = lane * 4 + j;
      seq[b * 896 + seq_off + n] = (_Float16)Emb[best * 128 + n];
    }
  }
  if (lane < NL) pin[b * 544 + oh_off + lane] = (_Float16)((lane == best) ? 1.f : 0.f);
  if (lane == 0) {
    idxbuf[b * 4 + slot] = best;
    if (slot == 3) {
      int s = idxbuf[b * 4] * 64 + idxbuf[b * 4 + 1] * 16 + idxbuf[b * 4 + 2] * 4 + best;
      out_sidx[b] = (float)s;
    }
  }
}

__global__ __launch_bounds__(256) void ple_mm_tail_k(const _Float16* __restrict__ mmhid,
                                                     const float* __restrict__ maeW2,
                                                     const float* __restrict__ maeB2,
                                                     const float* __restrict__ mfeW2,
                                                     const float* __restrict__ mfeB2,
                                                     float* __restrict__ out_mae,
                                                     float* __restrict__ out_mfe) {
  int wave = threadIdx.x >> 5, lane = threadIdx.x & 31;
  size_t b = (size_t)blockIdx.x * 8 + wave;
  float pa = 0.f, pf = 0.f;
#pragma unroll
  for (int j = 0; j < 2; ++j) {
    int n = lane * 2 + j;
    pa += (float)mmhid[b * 128 + n] * maeW2[n];
    pf += (float)mmhid[b * 128 + 64 + n] * mfeW2[n];
  }
  pa = wred(pa);
  pf = wred(pf);
  if (lane == 0) {
    out_mae[b] = pa + maeB2[0];
    out_mfe[b] = pf + mfeB2[0];
  }
}

__global__ __launch_bounds__(256) void ple_conf_tail_k(const _Float16* __restrict__ chid,
                                                       const float* __restrict__ cW2,
                                                       const float* __restrict__ cB2,
                                                       float* __restrict__ out_conf) {
  int wave = threadIdx.x >> 5, lane = threadIdx.x & 31;
  size_t b = (size_t)blockIdx.x * 8 + wave;
  float p = (float)chid[b * 32 + lane] * cW2[lane];
  p = wred(p);
  if (lane == 0) out_conf[b] = 1.f / (1.f + expf(-(p + cB2[0])));
}

// ---------------- launcher ----------------

extern "C" void kernel_launch(void* const* d_in, const int* in_sizes, int n_in,
                              void* d_out, int out_size, void* d_ws, size_t ws_size,
                              hipStream_t stream) {
  // input indices (setup_inputs dict order, params flattened in key order)
  enum {
    I_feat = 0, I_acct, I_idx,
    I_eW1, I_eB1, I_eW2, I_eB2, I_eW3, I_eB3, I_eLg, I_eLb,
    I_aW, I_aB, I_qW, I_qB, I_kW, I_kB, I_fW, I_fB, I_fLg, I_fLb,
    I_dW1, I_dB1, I_dW2, I_dB2, I_dE,
    I_rW1, I_rB1, I_rW2, I_rB2, I_rE,
    I_tW1, I_tB1, I_tW2, I_tB2, I_tE,
    I_rrW1, I_rrB1, I_rrW2, I_rrB2,
    I_maeW1, I_maeB1, I_maeW2, I_maeB2,
    I_mfeW1, I_mfeB1, I_mfeW2, I_mfeB2,
    I_cW1, I_cB1, I_cW2, I_cB2
  };
  auto F = [&](int i) { return (const float*)d_in[i]; };

  size_t off = 0;
  auto alloc = [&](size_t bytes) -> void* {
    void* p = (char*)d_ws + off;
    off = (off + bytes + 255) & ~(size_t)255;
    return p;
  };
  // activations
  _Float16* eoacct = (_Float16*)alloc((size_t)Bn * 1152 * 2);  // [eo(1024) | acct(128)]
  _Float16* qh     = (_Float16*)alloc((size_t)Bn * 256 * 2);
  _Float16* kkh    = (_Float16*)alloc((size_t)Bn * 1024 * 2);
  _Float16* gacct  = (_Float16*)alloc((size_t)Bn * 384 * 2);   // [gated(256) | acct(128)]
  _Float16* seq    = (_Float16*)alloc((size_t)Bn * 896 * 2);   // [fused | de | re | te]
  _Float16* pin    = (_Float16*)alloc((size_t)Bn * 544 * 2);   // [fused | oh(14) | pad]
  _Float16* hid    = (_Float16*)alloc((size_t)Bn * 256 * 2);   // reused across 4 heads
  _Float16* mmhid  = (_Float16*)alloc((size_t)Bn * 128 * 2);
  _Float16* chid   = (_Float16*)alloc((size_t)Bn * 32 * 2);
  int*      idxbuf = (int*)alloc((size_t)Bn * 4 * 4);
  // f16 weights
  _Float16* eW1h  = (_Float16*)alloc(131072 * 2);
  _Float16* eW2h  = (_Float16*)alloc(1048576 * 2);
  _Float16* eW3h  = (_Float16*)alloc(524288 * 2);
  _Float16* qWh   = (_Float16*)alloc(294912 * 2);
  _Float16* kWh   = (_Float16*)alloc(262144 * 2);
  _Float16* fWh   = (_Float16*)alloc(196608 * 2);
  _Float16* dW1h  = (_Float16*)alloc(131072 * 2);
  _Float16* rW1h  = (_Float16*)alloc(163840 * 2);
  _Float16* tW1h  = (_Float16*)alloc(196608 * 2);
  _Float16* rrW1h = (_Float16*)alloc(229376 * 2);
  _Float16* cW1h  = (_Float16*)alloc(16384 * 2);
  _Float16* mmW1h = (_Float16*)alloc(544 * 128 * 2);
  float*    mmB   = (float*)alloc(128 * 4);

  auto pack = [&](int i, _Float16* dst, int n) {
    ple_pack_f16_k<<<(n + 255) / 256, 256, 0, stream>>>(F(i), dst, n);
  };
  pack(I_eW1, eW1h, 131072);
  pack(I_eW2, eW2h, 1048576);
  pack(I_eW3, eW3h, 524288);
  pack(I_qW, qWh, 294912);
  pack(I_kW, kWh, 262144);
  pack(I_fW, fWh, 196608);
  pack(I_dW1, dW1h, 131072);
  pack(I_rW1, rW1h, 163840);
  pack(I_tW1, tW1h, 196608);
  pack(I_rrW1, rrW1h, 229376);
  pack(I_cW1, cW1h, 16384);
  ple_packmm_k<<<(544 * 128 + 255) / 256, 256, 0, stream>>>(F(I_maeW1), F(I_mfeW1),
                                                            F(I_maeB1), F(I_mfeB1), mmW1h, mmB);

  float* out = (float*)d_out;
  float* out_dl   = out;                    // B x 2
  float* out_rl   = out + 2 * (size_t)Bn;   // B x 4
  float* out_tl   = out + 6 * (size_t)Bn;
  float* out_rrl  = out + 10 * (size_t)Bn;
  float* out_gw   = out + 14 * (size_t)Bn;  // B x 4
  float* out_mae  = out + 18 * (size_t)Bn;
  float* out_mfe  = out + 19 * (size_t)Bn;
  float* out_conf = out + 20 * (size_t)Bn;
  float* out_sidx = out + 21 * (size_t)Bn;

  // expert stack (fused, WMMA-heavy) -> eoacct[:, 0:1024]
  ple_expert_k<<<dim3(Bn / 32, 4), 256, 0, stream>>>(
      F(I_feat), (const int*)d_in[I_idx], eW1h, F(I_eB1), eW2h, F(I_eB2), eW3h, F(I_eB3),
      F(I_eLg), F(I_eLb), eoacct);
  // acct MLP -> eoacct[:,1024:1152] and gacct[:,256:384]
  ple_acct_k<<<Bn * 32 / 256, 256, 0, stream>>>(F(I_acct), F(I_aW), F(I_aB), eoacct, gacct);
  // q = [flat|acct] @ qW + qB
  ple_gemm_k<1152, 256, 0><<<Bn / 32, 256, 0, stream>>>(eoacct, 1152, qWh, F(I_qB), qh, 256);
  // kk per expert
  for (int e = 0; e < 4; ++e)
    ple_gemm_k<256, 256, 0><<<Bn / 32, 256, 0, stream>>>(
        eoacct + e * 256, 1152, kWh + (size_t)e * 65536, F(I_kB) + e * 256, kkh + e * 256, 1024);
  // gating softmax + mix -> gacct[:,0:256], gw out
  ple_gate_k<<<Bn / 8, 256, 0, stream>>>(qh, kkh, eoacct, gacct, out_gw);
  // fusion layer + LN -> seq[:,0:512], pin[:,0:512] (+zero pad)
  ple_fuse_ln_k<<<Bn / 16, 256, 0, stream>>>(gacct, fWh, F(I_fB), F(I_fLg), F(I_fLb), seq, pin);

  // sequential decision heads
  ple_gemm_k<512, 256, 1><<<Bn / 32, 256, 0, stream>>>(seq, 896, dW1h, F(I_dB1), hid, 256);
  ple_head_tail_k<<<Bn / 8, 256, 0, stream>>>(hid, F(I_dW2), F(I_dB2), 2, F(I_dE), seq, 512,
                                              pin, 512, idxbuf, 0, out_dl, nullptr);
  ple_gemm_k<640, 256, 1><<<Bn / 32, 256, 0, stream>>>(seq, 896, rW1h, F(I_rB1), hid, 256);
  ple_head_tail_k<<<Bn / 8, 256, 0, stream>>>(hid, F(I_rW2), F(I_rB2), 4, F(I_rE), seq, 640,
                                              pin, 514, idxbuf, 1, out_rl, nullptr);
  ple_gemm_k<768, 256, 1><<<Bn / 32, 256, 0, stream>>>(seq, 896, tW1h, F(I_tB1), hid, 256);
  ple_head_tail_k<<<Bn / 8, 256, 0, stream>>>(hid, F(I_tW2), F(I_tB2), 4, F(I_tE), seq, 768,
                                              pin, 518, idxbuf, 2, out_tl, nullptr);
  ple_gemm_k<896, 256, 1><<<Bn / 32, 256, 0, stream>>>(seq, 896, rrW1h, F(I_rrB1), hid, 256);
  ple_head_tail_k<<<Bn / 8, 256, 0, stream>>>(hid, F(I_rrW2), F(I_rrB2), 4, nullptr, seq, 0,
                                              pin, 522, idxbuf, 3, out_rrl, out_sidx);

  // mae/mfe (combined N=128) and confidence
  ple_gemm_k<544, 128, 1><<<Bn / 32, 256, 0, stream>>>(pin, 544, mmW1h, mmB, mmhid, 128);
  ple_mm_tail_k<<<Bn / 8, 256, 0, stream>>>(mmhid, F(I_maeW2), F(I_maeB2), F(I_mfeW2),
                                            F(I_mfeB2), out_mae, out_mfe);
  ple_gemm_k<512, 32, 1><<<Bn / 32, 256, 0, stream>>>(seq, 896, cW1h, F(I_cB1), chid, 32);
  ple_conf_tail_k<<<Bn / 8, 256, 0, stream>>>(chid, F(I_cW2), F(I_cB2), out_conf);
  (void)in_sizes; (void)n_in; (void)out_size; (void)ws_size;
}
